// SGATFunc_63660005261791
// MI455X (gfx1250) — matrix operand order repaired and checked by reading, not compiled, not run on hardware
//
#include <hip/hip_runtime.h>

#define B 32
#define N 1024
#define D 64
#define NWAVES 8

typedef __bf16 bf16_t;
typedef bf16_t v16bf __attribute__((ext_vector_type(16)));
typedef float  v8f   __attribute__((ext_vector_type(8)));

// Raw hardware reciprocal (v_rcp_f32, ~1 ulp) -- skips Newton refinement.
__device__ __forceinline__ float rcpf_(float x) {
    return __builtin_amdgcn_rcpf(x);
}

__device__ __forceinline__ float sigmoidf_(float x) {
    return rcpf_(1.0f + __expf(-x));
}

// tanh(x) = 1 - 2/(exp(2x)+1); exact identity, saturates correctly at +-1.
__device__ __forceinline__ float fast_tanhf_(float x) {
    return 1.0f - 2.0f * rcpf_(__expf(2.0f * x) + 1.0f);
}

// 16-bit A-matrix (16x32) k order chosen here (contiguous, B uses the same; not AMD's table order): lanes 0-15 hold K {0..15},
// lanes 16-31 hold K {16..31}; element idx 0..15 within v16bf.
__device__ __forceinline__ int a_koff(int half, int idx) {
    return half * 16 + idx;
}

// Shared memory for the fused kernel (~104 KB of the 320 KB WGP budget).
struct SharedBuf {
    float  ef[16][N + 4];     // E_f rows (f32, padded: stride 1028 -> conflict-free A-gather)
    bf16_t nh[16][N + 8];     // N_hat rows (bf16, stride 1032 halves)
    float  vs[16][33];        // staged vs tile (16x32, padded)
    float  e1c[32];           // staged e1 chunk
    float  red[16][NWAVES];   // cross-wave reduction scratch
    float  rowv[16];          // per-row max / sum
    float  outb[16][D + 4];   // output tile reduction buffer
};

// Computes 16 rows (I0..I0+15) of softmax scores for one batch.
// On return: acc[t][r] holds exp(e - rowmax); rinv[r] = 1/rowsum.
// C-layout: acc[t][r] is element (row = r + 8*half, col = wave*128 + t*16 + m).
__device__ void compute_attn(const float* __restrict__ vsM,
                             const float* __restrict__ bsM,
                             const float* __restrict__ e1b,
                             const float* __restrict__ e2b,
                             int I0, int tid, v8f acc[8], float rinv[8],
                             SharedBuf& sh) {
    const int wave = tid >> 5;
    const int lane = tid & 31;
    const int m = lane & 15;
    const int half = lane >> 4;

    float e2v[8];
#pragma unroll
    for (int t = 0; t < 8; ++t) {
        e2v[t] = e2b[wave * 128 + t * 16 + m];
#pragma unroll
        for (int r = 0; r < 8; ++r) acc[t][r] = 0.0f;
    }

    for (int k0 = 0; k0 < N; k0 += 32) {
        __syncthreads();
        for (int i = tid; i < 512; i += 256) {
            sh.vs[i >> 5][i & 31] = vsM[(size_t)(I0 + (i >> 5)) * N + k0 + (i & 31)];
        }
        if (tid < 32) sh.e1c[tid] = e1b[k0 + tid];
        __syncthreads();

        // A = vs[I0:I0+16, k0:k0+32] in bf16 A-layout (same for all waves)
        v16bf A;
#pragma unroll
        for (int i = 0; i < 16; ++i)
            A[i] = (bf16_t)sh.vs[m][a_koff(half, i)];

        if (k0 + 32 < N)
            __builtin_prefetch(bsM + (size_t)(k0 + 32 + half * 16) * N + wave * 128 + m, 0, 0);

#pragma unroll
        for (int t = 0; t < 8; ++t) {
            // B = sigmoid(e1[k] * e2[col] + bs[k, col]), built directly in B-layout:
            // lane col = (lane&15), K half = (lane>>4)*16, 16 K values per lane.
            const int col = wave * 128 + t * 16 + m;
            const float* bsp = bsM + (size_t)(k0 + half * 16) * N + col;
            v16bf Bm;
#pragma unroll
            for (int i = 0; i < 16; ++i) {
                float x = sh.e1c[half * 16 + i] * e2v[t] + bsp[(size_t)i * N];
                Bm[i] = (bf16_t)sigmoidf_(x);
            }
            acc[t] = __builtin_amdgcn_wmma_f32_16x16x32_bf16(
                false, A, false, Bm, (short)0, acc[t], false, false);
        }
    }

    // ---- row softmax: row (r + 8*half) lives in 16 lanes of this half ----
    float pm[8];
#pragma unroll
    for (int r = 0; r < 8; ++r) {
        float v = acc[0][r];
#pragma unroll
        for (int t = 1; t < 8; ++t) v = fmaxf(v, acc[t][r]);
#pragma unroll
        for (int s = 1; s < 16; s <<= 1) v = fmaxf(v, __shfl_xor(v, s, 32));
        pm[r] = v;
    }
    __syncthreads();
    if (m == 0) {
#pragma unroll
        for (int r = 0; r < 8; ++r) sh.red[r + 8 * half][wave] = pm[r];
    }
    __syncthreads();
    if (tid < 16) {
        float v = sh.red[tid][0];
#pragma unroll
        for (int w = 1; w < NWAVES; ++w) v = fmaxf(v, sh.red[tid][w]);
        sh.rowv[tid] = v;
    }
    __syncthreads();

    float ps[8];
#pragma unroll
    for (int r = 0; r < 8; ++r) {
        float rm = sh.rowv[r + 8 * half];
        float s = 0.0f;
#pragma unroll
        for (int t = 0; t < 8; ++t) {
            acc[t][r] = __expf(acc[t][r] - rm);
            s += acc[t][r];
        }
#pragma unroll
        for (int sft = 1; sft < 16; sft <<= 1) s += __shfl_xor(s, sft, 32);
        ps[r] = s;
    }
    __syncthreads();
    if (m == 0) {
#pragma unroll
        for (int r = 0; r < 8; ++r) sh.red[r + 8 * half][wave] = ps[r];
    }
    __syncthreads();
    if (tid < 16) {
        float v = 0.0f;
#pragma unroll
        for (int w = 0; w < NWAVES; ++w) v += sh.red[tid][w];
        sh.rowv[tid] = v;
    }
    __syncthreads();
#pragma unroll
    for (int r = 0; r < 8; ++r) rinv[r] = rcpf_(sh.rowv[r + 8 * half]);
}

// ---------------- prep kernel: e1/e2 dots, W_hat, sigmoid(alpha), A_mix ------
__global__ void prep_kernel(const float* __restrict__ xr,
                            const float* __restrict__ w,
                            const float* __restrict__ dvec,
                            const float* __restrict__ alpha,
                            const float* __restrict__ fw1, const float* __restrict__ fw2,
                            const float* __restrict__ gw1, const float* __restrict__ gw2,
                            const float* __restrict__ conv_w,
                            const float* __restrict__ adj,
                            float* __restrict__ e1f, float* __restrict__ e2f,
                            float* __restrict__ e1g, float* __restrict__ e2g,
                            float* __restrict__ Wh, float* __restrict__ sal,
                            float* __restrict__ amix) {
    const size_t gid = (size_t)blockIdx.x * blockDim.x + threadIdx.x;
    const size_t T0 = (size_t)4 * B * N;
    const size_t T1 = T0 + (size_t)D * D;
    const size_t T2 = T1 + N;
    const size_t T3 = T2 + (size_t)N * N;
    if (gid < T0) {
        int which = (int)(gid / ((size_t)B * N));
        int bn = (int)(gid % ((size_t)B * N));
        const float* wv = (which == 0) ? fw1 : (which == 1) ? fw2 : (which == 2) ? gw1 : gw2;
        const float* xp = xr + (size_t)bn * D;
        float s = 0.0f;
#pragma unroll 8
        for (int dd = 0; dd < D; ++dd) s += xp[dd] * wv[dd];
        float* outp = (which == 0) ? e1f : (which == 1) ? e2f : (which == 2) ? e1g : e2g;
        outp[bn] = s;
    } else if (gid < T1) {
        int ij = (int)(gid - T0);
        int i = ij >> 6, j = ij & 63;
        float s = 0.0f;
        for (int kd = 0; kd < D; ++kd) {
            float dc = fminf(fmaxf(dvec[kd], 0.0f), 1.0f);
            s += w[i * D + kd] * dc * w[j * D + kd];
        }
        if (i == j) s -= 1.0f;
        Wh[ij] = s;
    } else if (gid < T2) {
        int i = (int)(gid - T1);
        sal[i] = 1.0f / (1.0f + __expf(-alpha[i]));  // exact: tiny array, keep precise
    } else if (gid < T3) {
        size_t ij = gid - T2;
        int i = (int)(ij >> 10), j = (int)(ij & 1023);
        float cw0 = conv_w[0], cw1 = conv_w[1];
        float v = cw0 * adj[ij] + cw1 * adj[(size_t)N * N + ij];
        if (i == j) v -= (cw0 + cw1);
        amix[ij] = v;
    }
}

// ---------------- xw = x @ W_hat via WMMA (K = 64) ---------------------------
__global__ void __launch_bounds__(256) xw_kernel(const float* __restrict__ xr,
                                                 const float* __restrict__ Wh,
                                                 float* __restrict__ xw) {
    __shared__ float s_wh[D][D + 1];
    const int tid = threadIdx.x;
    for (int i = tid; i < D * D; i += 256) s_wh[i >> 6][i & 63] = Wh[i];
    __syncthreads();

    const int wave = tid >> 5, lane = tid & 31;
    const int m = lane & 15, half = lane >> 4;
    const size_t R0 = ((size_t)blockIdx.x * NWAVES + wave) * 16;  // base row in B*N

    v8f o[4];
#pragma unroll
    for (int nt = 0; nt < 4; ++nt)
#pragma unroll
        for (int r = 0; r < 8; ++r) o[nt][r] = 0.0f;

#pragma unroll
    for (int k0 = 0; k0 < D; k0 += 32) {
        v16bf A;
#pragma unroll
        for (int i = 0; i < 16; ++i)
            A[i] = (bf16_t)xr[(R0 + m) * D + k0 + a_koff(half, i)];
#pragma unroll
        for (int nt = 0; nt < 4; ++nt) {
            const int col = nt * 16 + m;
            v16bf Bm;
#pragma unroll
            for (int i = 0; i < 16; ++i)
                Bm[i] = (bf16_t)s_wh[k0 + half * 16 + i][col];
            o[nt] = __builtin_amdgcn_wmma_f32_16x16x32_bf16(
                false, A, false, Bm, (short)0, o[nt], false, false);
        }
    }
#pragma unroll
    for (int nt = 0; nt < 4; ++nt)
#pragma unroll
        for (int r = 0; r < 8; ++r)
            xw[(R0 + r + 8 * half) * D + nt * 16 + m] = o[nt][r];
}

// ---------------- fused attention + message passing kernel -------------------
__global__ void __launch_bounds__(256) fused_kernel(
    const float* __restrict__ xr, const float* __restrict__ x0,
    const float* __restrict__ fvs, const float* __restrict__ fbs,
    const float* __restrict__ gvs, const float* __restrict__ gbs,
    const float* __restrict__ e1f, const float* __restrict__ e2f,
    const float* __restrict__ e1g, const float* __restrict__ e2g,
    const float* __restrict__ amix, const float* __restrict__ xw,
    const float* __restrict__ sal, const float* __restrict__ conv_b,
    float* __restrict__ fout, float* __restrict__ gout) {
    __shared__ SharedBuf sh;
    const int b = blockIdx.y;
    const int I0 = blockIdx.x * 16;
    const int tid = threadIdx.x;
    const int wave = tid >> 5, lane = tid & 31;
    const int m = lane & 15, half = lane >> 4;

    v8f acc[8];
    float rinv[8];

    // ---- Phase 1: E_f rows (softmax attention, f params) -> LDS f32 ----
    compute_attn(fvs, fbs, e1f + (size_t)b * N, e2f + (size_t)b * N, I0, tid, acc, rinv, sh);
#pragma unroll
    for (int t = 0; t < 8; ++t)
#pragma unroll
        for (int r = 0; r < 8; ++r)
            sh.ef[r + 8 * half][wave * 128 + t * 16 + m] = acc[t][r] * rinv[r];

    // ---- Phase 2: N_att rows (g params); N_hat = N_att / (E_f + eps) -> LDS bf16 ----
    compute_attn(gvs, gbs, e1g + (size_t)b * N, e2g + (size_t)b * N, I0, tid, acc, rinv, sh);
    __syncthreads();
#pragma unroll
    for (int t = 0; t < 8; ++t)
#pragma unroll
        for (int r = 0; r < 8; ++r) {
            int row = r + 8 * half, col = wave * 128 + t * 16 + m;
            float ef = sh.ef[row][col];
            sh.nh[row][col] = (bf16_t)((acc[t][r] * rinv[r]) * rcpf_(ef + 1e-5f));
        }
    __syncthreads();

    const float* xrb = xr + (size_t)b * N * D;

    // ---- Phase 3: g = tanh(N_hat @ x)  (waves split K, LDS f32 reduce) ----
    v8f og[4];
#pragma unroll
    for (int nt = 0; nt < 4; ++nt)
#pragma unroll
        for (int r = 0; r < 8; ++r) og[nt][r] = 0.0f;

    for (int kk = 0; kk < 128; kk += 32) {
        const int kbase = wave * 128 + kk;
        v16bf A;
#pragma unroll
        for (int i = 0; i < 16; ++i)
            A[i] = sh.nh[m][kbase + a_koff(half, i)];
#pragma unroll
        for (int nt = 0; nt < 4; ++nt) {
            const int col = nt * 16 + m;
            const float* xp = xrb + (size_t)(kbase + half * 16) * D + col;
            v16bf Bm;
#pragma unroll
            for (int i = 0; i < 16; ++i) Bm[i] = (bf16_t)xp[(size_t)i * D];
            og[nt] = __builtin_amdgcn_wmma_f32_16x16x32_bf16(
                false, A, false, Bm, (short)0, og[nt], false, false);
        }
    }
    for (int i = tid; i < 16 * D; i += 256) sh.outb[i >> 6][i & 63] = 0.0f;
    __syncthreads();
#pragma unroll
    for (int nt = 0; nt < 4; ++nt)
#pragma unroll
        for (int r = 0; r < 8; ++r)
            atomicAdd(&sh.outb[r + 8 * half][nt * 16 + m], og[nt][r]);
    __syncthreads();
    for (int i = tid; i < 16 * D; i += 256) {
        int r = i >> 6, c = i & 63;
        gout[((size_t)b * N + I0 + r) * D + c] = fast_tanhf_(sh.outb[r][c]);
    }
    __syncthreads();

    // ---- Phase 4: f = tanh((E_f - I + A_mix) @ x + xw + x0*sig(alpha) + cb) ----
    v8f of[4];
#pragma unroll
    for (int nt = 0; nt < 4; ++nt)
#pragma unroll
        for (int r = 0; r < 8; ++r) of[nt][r] = 0.0f;

    for (int kk = 0; kk < 128; kk += 32) {
        const int kbase = wave * 128 + kk;
        const int row = I0 + m;
        const float* amp = amix + (size_t)row * N + kbase;
        v16bf A;
#pragma unroll
        for (int i = 0; i < 16; ++i) {
            int ko = a_koff(half, i);
            float v = sh.ef[m][kbase + ko] + amp[ko];
            if (kbase + ko == row) v -= 1.0f;
            A[i] = (bf16_t)v;
        }
#pragma unroll
        for (int nt = 0; nt < 4; ++nt) {
            const int col = nt * 16 + m;
            const float* xp = xrb + (size_t)(kbase + half * 16) * D + col;
            v16bf Bm;
#pragma unroll
            for (int i = 0; i < 16; ++i) Bm[i] = (bf16_t)xp[(size_t)i * D];
            of[nt] = __builtin_amdgcn_wmma_f32_16x16x32_bf16(
                false, A, false, Bm, (short)0, of[nt], false, false);
        }
    }
    for (int i = tid; i < 16 * D; i += 256) sh.outb[i >> 6][i & 63] = 0.0f;
    __syncthreads();
#pragma unroll
    for (int nt = 0; nt < 4; ++nt)
#pragma unroll
        for (int r = 0; r < 8; ++r)
            atomicAdd(&sh.outb[r + 8 * half][nt * 16 + m], of[nt][r]);
    __syncthreads();
    const float cb = conv_b[0];
    for (int i = tid; i < 16 * D; i += 256) {
        int r = i >> 6, c = i & 63;
        size_t idx = ((size_t)b * N + I0 + r) * D + c;
        float v = sh.outb[r][c] + xw[idx] + x0[idx] * sal[I0 + r] + cb;
        fout[idx] = fast_tanhf_(v);
    }
}

extern "C" void kernel_launch(void* const* d_in, const int* in_sizes, int n_in,
                              void* d_out, int out_size, void* d_ws, size_t ws_size,
                              hipStream_t stream) {
    const float* xr    = (const float*)d_in[0];
    const float* x0    = (const float*)d_in[1];
    const float* w     = (const float*)d_in[2];
    const float* dvec  = (const float*)d_in[3];
    const float* alpha = (const float*)d_in[4];
    const float* fw1   = (const float*)d_in[5];
    const float* fw2   = (const float*)d_in[6];
    const float* fvs   = (const float*)d_in[7];
    const float* fbs   = (const float*)d_in[8];
    const float* gw1   = (const float*)d_in[9];
    const float* gw2   = (const float*)d_in[10];
    const float* gvs   = (const float*)d_in[11];
    const float* gbs   = (const float*)d_in[12];
    const float* cw    = (const float*)d_in[13];
    const float* cb    = (const float*)d_in[14];
    const float* adj   = (const float*)d_in[15];

    float* fout = (float*)d_out;
    float* gout = fout + (size_t)B * N * D;

    // workspace layout (~13.1 MB of f32)
    float* wsf  = (float*)d_ws;
    float* amix = wsf;                      // N*N
    float* xw   = amix + (size_t)N * N;     // B*N*D
    float* Wh   = xw + (size_t)B * N * D;   // D*D
    float* e1f  = Wh + (size_t)D * D;       // B*N each
    float* e2f  = e1f + (size_t)B * N;
    float* e1g  = e2f + (size_t)B * N;
    float* e2g  = e1g + (size_t)B * N;
    float* sal  = e2g + (size_t)B * N;      // N

    {
        size_t total = (size_t)4 * B * N + (size_t)D * D + N + (size_t)N * N;
        int blocks = (int)((total + 255) / 256);
        prep_kernel<<<blocks, 256, 0, stream>>>(xr, w, dvec, alpha, fw1, fw2, gw1, gw2,
                                                cw, adj, e1f, e2f, e1g, e2g, Wh, sal, amix);
    }
    xw_kernel<<<(B * N) / (16 * NWAVES), 256, 0, stream>>>(xr, Wh, xw);
    fused_kernel<<<dim3(N / 16, B), 256, 0, stream>>>(
        xr, x0, fvs, fbs, gvs, gbs, e1f, e2f, e1g, e2g, amix, xw, sal, cb, fout, gout);
}
